// InvariantPointAttention_59287728554231
// MI455X (gfx1250) — matrix operand order repaired and checked by reading, not compiled, not run on hardware
//
#include <hip/hip_runtime.h>

#define DEV __device__ __forceinline__

typedef __attribute__((ext_vector_type(16))) __bf16 bf16x16;
typedef __attribute__((ext_vector_type(8)))  float  f32x8;

DEV __bf16 f2bf(float f) {
  unsigned u = __builtin_bit_cast(unsigned, f);
  unsigned r = u + 0x7FFFu + ((u >> 16) & 1u);   // round-to-nearest-even
  unsigned short h = (unsigned short)(r >> 16);
  return __builtin_bit_cast(__bf16, h);
}

DEV __bf16 bfbits(unsigned short h) { return __builtin_bit_cast(__bf16, h); }

DEV unsigned pack2(float a, float b) {
  unsigned ha = (unsigned)__builtin_bit_cast(unsigned short, f2bf(a));
  unsigned hb = (unsigned)__builtin_bit_cast(unsigned short, f2bf(b));
  return ha | (hb << 16);
}

// A fragment: 16x32 bf16, row-major source with leading dim lda.
// lanes 0-15 -> M=lane, K=0..15;
// lanes 16-31 -> M=lane-16, K=16..31.
DEV bf16x16 load_a_frag(const __bf16* A, int lda) {
  int lane = threadIdx.x & 31;
  const __bf16* row = A + (size_t)(lane & 15) * lda;
  int kb = (lane < 16) ? 0 : 16;
  bf16x16 f;
#pragma unroll
  for (int v = 0; v < 8; ++v) {
    int k = kb + 2 * v;
    f[2 * v]     = row[k];
    f[2 * v + 1] = row[k + 1];
  }
  return f;
}

// B fragment: 32x16 bf16 from a K-major (B-transposed, [n][k]) source.
// lanes 0-15 -> N=lane, K=0..15; lanes 16-31 -> N=lane-16, K=16..31.
// Caller guarantees 32-byte alignment -> single 32B vector load (2 x b128).
DEV bf16x16 load_b_frag(const __bf16* Bt, int ldb) {
  int lane = threadIdx.x & 31;
  const __bf16* row = Bt + (size_t)(lane & 15) * ldb + ((lane < 16) ? 0 : 16);
  return *(const bf16x16*)row;
}

DEV f32x8 wmma_bf16(bf16x16 a, bf16x16 b, f32x8 c) {
  return __builtin_amdgcn_wmma_f32_16x16x32_bf16(false, a, false, b, (short)0, c,
                                                 false, false);
}

// ---------------------------------------------------------------- conversion
__global__ void cvt_f32_bf16_v4(const float4* __restrict__ in,
                                uint2* __restrict__ out, int n4) {
  int i = blockIdx.x * blockDim.x + threadIdx.x;
  if (i < n4) {
    float4 v = in[i];
    out[i] = make_uint2(pack2(v.x, v.y), pack2(v.z, v.w));
  }
}

// --------------------------------------------------- generic bf16 WMMA GEMM
// C[M,N] = A[M,K] x B[K,N], A/B bf16 row-major, C f32.
// Block tile 128x64, 8 waves = 4x2 grid of 32x32 wave tiles (4 WMMA per wave
// per K-step, 2x fragment reuse). Double-buffered LDS, b128 staging loads.
// Guarded N tail (for N=96).
__global__ __launch_bounds__(256) void gemm_bf16_kernel(
    const __bf16* __restrict__ A, const __bf16* __restrict__ B,
    float* __restrict__ C, int M, int N, int K) {
  __shared__ __bf16 As[2][128][40];   // stride 40: 16B-aligned vector stores
  __shared__ __bf16 Bt[2][64][48];    // stride 48: 32B-aligned fragment rows
  const int tid  = threadIdx.x;
  const int wave = tid >> 5;
  const int mw = (wave & 3) * 32;
  const int nw = (wave >> 2) * 32;
  const int bm = blockIdx.x * 128;
  const int bn = blockIdx.y * 64;
  const __bf16* Abase = A + (size_t)bm * K;

  const int ar0 = (tid) >> 2,        ac0 = ((tid) & 3) * 8;
  const int ar1 = (tid + 256) >> 2,  ac1 = ((tid + 256) & 3) * 8;
  const int bkr = tid >> 3,          bnc = (tid & 7) * 8;
  const bool bval = (bn + bnc + 8 <= N);

  uint4 a0r, a1r, b0r = make_uint4(0, 0, 0, 0);

  auto stage_load = [&](int k0) {
    a0r = *(const uint4*)(Abase + (size_t)ar0 * K + k0 + ac0);
    a1r = *(const uint4*)(Abase + (size_t)ar1 * K + k0 + ac1);
    if (bval) b0r = *(const uint4*)(B + (size_t)(k0 + bkr) * N + bn + bnc);
  };
  auto stage_store = [&](int bufi) {
    *(uint4*)&As[bufi][ar0][ac0] = a0r;
    *(uint4*)&As[bufi][ar1][ac1] = a1r;
    unsigned w[4] = {b0r.x, b0r.y, b0r.z, b0r.w};
#pragma unroll
    for (int i = 0; i < 4; ++i) {
      unsigned lo = bval ? (w[i] & 0xffffu) : 0u;
      unsigned hi = bval ? (w[i] >> 16) : 0u;
      Bt[bufi][bnc + 2 * i][bkr]     = bfbits((unsigned short)lo);
      Bt[bufi][bnc + 2 * i + 1][bkr] = bfbits((unsigned short)hi);
    }
  };

  f32x8 acc00 = {}, acc01 = {}, acc10 = {}, acc11 = {};
  stage_load(0);
  stage_store(0);
  __syncthreads();
  int buf = 0;
  for (int k0 = 0; k0 < K; k0 += 32) {
    bool has_next = (k0 + 32 < K);
    if (has_next) stage_load(k0 + 32);        // overlap VMEM with WMMA below
    bf16x16 fa0 = load_a_frag(&As[buf][mw][0], 40);
    bf16x16 fa1 = load_a_frag(&As[buf][mw + 16][0], 40);
    bf16x16 fb0 = load_b_frag(&Bt[buf][nw][0], 48);
    bf16x16 fb1 = load_b_frag(&Bt[buf][nw + 16][0], 48);
    acc00 = wmma_bf16(fa0, fb0, acc00);
    acc01 = wmma_bf16(fa0, fb1, acc01);
    acc10 = wmma_bf16(fa1, fb0, acc10);
    acc11 = wmma_bf16(fa1, fb1, acc11);
    if (has_next) stage_store(buf ^ 1);
    __syncthreads();
    buf ^= 1;
  }

  int lane = tid & 31;
  int n  = lane & 15;
  int mb = (lane < 16) ? 0 : 8;
  const f32x8 accs[2][2] = {{acc00, acc01}, {acc10, acc11}};
#pragma unroll
  for (int i = 0; i < 2; ++i)
#pragma unroll
    for (int j = 0; j < 2; ++j) {
      int col = bn + nw + j * 16 + n;
      if (col < N) {
        float* Cp = C + (size_t)(bm + mw + i * 16 + mb) * N + col;
#pragma unroll
        for (int v = 0; v < 8; ++v) Cp[(size_t)v * N] = accs[i][j][v];
      }
    }
}

// ----------------------------------------- rigid transform + extended Q/K/V
// One thread per (b,l,h). Builds:
//   Qe[bh,l,96]  = [Q/8, sqrt(wc)*qp_g, 0...]
//   Ke[bh,l,96]  = [K,   sqrt(wc)*kp_g, 0...]
//   VeT[bh,80,l] = transpose of [V, vp_g, 0-pad]
//   Kbias[bh,l]  = -0.5*wc*sum|kp_g|^2   (column softmax bias)
__global__ void point_setup_kernel(
    const float* __restrict__ Qf, const float* __restrict__ Kf,
    const float* __restrict__ Vf, const float* __restrict__ Qpf,
    const float* __restrict__ Kpf, const float* __restrict__ Vpf,
    const float* __restrict__ R, const float* __restrict__ tr,
    const float* __restrict__ wC, __bf16* __restrict__ Qe,
    __bf16* __restrict__ Ke, __bf16* __restrict__ VeT,
    float* __restrict__ Kbias) {
  int idx = blockIdx.x * blockDim.x + threadIdx.x;
  if (idx >= 2 * 1024 * 8) return;
  int h = idx & 7;
  int n = idx >> 3;           // b*1024 + l
  int b = n >> 10;
  int l = n & 1023;
  int bh = b * 8 + h;
  float wc = log1pf(expf(wC[h]));
  float sw = sqrtf(wc);
  const float* Rm = R + (size_t)n * 9;
  const float* t  = tr + (size_t)n * 3;
  size_t qk = ((size_t)bh * 1024 + l) * 96;
  for (int d = 0; d < 64; ++d) {
    float qv = Qf[(size_t)n * 512 + h * 64 + d];
    float kv = Kf[(size_t)n * 512 + h * 64 + d];
    float vv = Vf[(size_t)n * 512 + h * 64 + d];
    Qe[qk + d] = f2bf(qv * 0.125f);              // 1/sqrt(64)
    Ke[qk + d] = f2bf(kv);
    VeT[((size_t)bh * 80 + d) * 1024 + l] = f2bf(vv);
  }
  float kb = 0.f;
  for (int p = 0; p < 4; ++p) {
    const float* qp = Qpf + (size_t)n * 96 + h * 12 + p * 3;
    const float* kp = Kpf + (size_t)n * 96 + h * 12 + p * 3;
    const float* vp = Vpf + (size_t)n * 96 + h * 12 + p * 3;
#pragma unroll
    for (int x = 0; x < 3; ++x) {
      float qg = Rm[x*3+0]*qp[0] + Rm[x*3+1]*qp[1] + Rm[x*3+2]*qp[2] + t[x];
      float kg = Rm[x*3+0]*kp[0] + Rm[x*3+1]*kp[1] + Rm[x*3+2]*kp[2] + t[x];
      float vg = Rm[x*3+0]*vp[0] + Rm[x*3+1]*vp[1] + Rm[x*3+2]*vp[2] + t[x];
      Qe[qk + 64 + p*3 + x] = f2bf(sw * qg);
      Ke[qk + 64 + p*3 + x] = f2bf(sw * kg);
      VeT[((size_t)bh * 80 + 64 + p*3 + x) * 1024 + l] = f2bf(vg);
      kb += kg * kg;
    }
  }
  for (int d = 76; d < 96; ++d) { Qe[qk + d] = f2bf(0.f); Ke[qk + d] = f2bf(0.f); }
  for (int d = 76; d < 80; ++d) VeT[((size_t)bh * 80 + d) * 1024 + l] = f2bf(0.f);
  Kbias[(size_t)bh * 1024 + l] = -0.5f * wc * kb;
}

// --------------------------------------------------------- attention kernel
// One block = one (b,h) and 32 query rows. Full 32x1024 f32 score tile in
// dynamic LDS (128KB of the 320KB WGP LDS), exact softmax, probabilities
// rewritten in-place as bf16, then O = P @ VeT via WMMA. Ke/VeT (~6MB total)
// stay L2-resident across all 512 blocks.
__global__ __launch_bounds__(256) void attention_kernel(
    const __bf16* __restrict__ Qe, const __bf16* __restrict__ Ke,
    const __bf16* __restrict__ VeT, const float* __restrict__ Kbias,
    float* __restrict__ O) {
  extern __shared__ char smem[];
  float*  Ss  = (float*)smem;                        // [32][1024] f32
  __bf16* Qes = (__bf16*)(smem + 32 * 1024 * 4);     // [32][96]
  const int bh   = blockIdx.x;
  const int m0   = blockIdx.y * 32;
  const int tid  = threadIdx.x;
  const int wave = tid >> 5;
  const int lane = tid & 31;
  {
    const uint2* src = (const uint2*)(Qe + ((size_t)bh * 1024 + m0) * 96);
    uint2* dst = (uint2*)Qes;
#pragma unroll
    for (int e = 0; e < 3; ++e) dst[tid + 256 * e] = src[tid + 256 * e];
  }
  __syncthreads();
  // ---- pass 1: S = Qe x Ke^T + c_j  (A-frags hoisted: 3 loads, 48 WMMA)
  {
    const int msub = wave & 1;
    bf16x16 aq[3];
#pragma unroll
    for (int ks = 0; ks < 3; ++ks)
      aq[ks] = load_a_frag(Qes + (size_t)(msub * 16) * 96 + ks * 32, 96);
    for (int nt = wave >> 1; nt < 64; nt += 4) {
      f32x8 acc = {};
#pragma unroll
      for (int ks = 0; ks < 3; ++ks) {
        bf16x16 bf = load_b_frag(Ke + ((size_t)bh * 1024 + nt * 16) * 96 + ks * 32, 96);
        acc = wmma_bf16(aq[ks], bf, acc);
      }
      int n  = lane & 15;
      int mb = (lane < 16) ? 0 : 8;
      float cj = Kbias[(size_t)bh * 1024 + nt * 16 + n];
#pragma unroll
      for (int v = 0; v < 8; ++v)
        Ss[(size_t)(msub * 16 + mb + v) * 1024 + nt * 16 + n] = acc[v] + cj;
    }
  }
  __syncthreads();
  // ---- pass 2: exact softmax, one wave per row; P written in-place as bf16
  for (int t = 0; t < 4; ++t) {
    int r = wave + 8 * t;
    float* row = Ss + (size_t)r * 1024;
    float ev[32];
    float mx = -3.4e38f;
#pragma unroll
    for (int i = 0; i < 32; ++i) {
      float x = row[lane + 32 * i];
      ev[i] = x;
      mx = fmaxf(mx, x);
    }
#pragma unroll
    for (int off = 16; off > 0; off >>= 1) mx = fmaxf(mx, __shfl_xor(mx, off, 32));
    float sum = 0.f;
#pragma unroll
    for (int i = 0; i < 32; ++i) { ev[i] = __expf(ev[i] - mx); sum += ev[i]; }
#pragma unroll
    for (int off = 16; off > 0; off >>= 1) sum += __shfl_xor(sum, off, 32);
    float inv = 1.f / sum;
    __bf16* prow = (__bf16*)row;
#pragma unroll
    for (int i = 0; i < 32; ++i) prow[lane + 32 * i] = f2bf(ev[i] * inv);
  }
  __syncthreads();
  // ---- pass 3: O = P @ VeT  (one A-frag per K-step feeds 1-2 WMMA;
  //      waves 0-1 carry the 5th N-tile; branch is wave-uniform so EXEC
  //      remains all-ones for WMMA)
  {
    int msub = wave & 1;
    int nt0  = wave >> 1;                 // 0..3
    bool two = (nt0 == 0);                // waves 0,1 also do nt=4
    f32x8 acc0 = {}, acc1 = {};
    for (int kt = 0; kt < 64; ++kt) {
      bf16x16 af = load_a_frag((__bf16*)smem + (size_t)(msub * 16) * 2048 + kt * 32, 2048);
      bf16x16 b0 = load_b_frag(VeT + ((size_t)bh * 80 + nt0 * 16) * 1024 + kt * 32, 1024);
      acc0 = wmma_bf16(af, b0, acc0);
      if (two) {
        bf16x16 b1 = load_b_frag(VeT + ((size_t)bh * 80 + 64) * 1024 + kt * 32, 1024);
        acc1 = wmma_bf16(af, b1, acc1);
      }
    }
    int n  = lane & 15;
    int mb = (lane < 16) ? 0 : 8;
#pragma unroll
    for (int v = 0; v < 8; ++v)
      O[((size_t)bh * 1024 + m0 + msub * 16 + mb + v) * 80 + nt0 * 16 + n] = acc0[v];
    if (two) {
#pragma unroll
      for (int v = 0; v < 8; ++v)
        O[((size_t)bh * 1024 + m0 + msub * 16 + mb + v) * 80 + 64 + n] = acc1[v];
    }
  }
}

// --------------------------------- inverse transform + concat -> Y (bf16)
__global__ void build_y_kernel(const float* __restrict__ O,
                               const float* __restrict__ R,
                               const float* __restrict__ tr,
                               __bf16* __restrict__ Y) {
  int idx = blockIdx.x * blockDim.x + threadIdx.x;
  if (idx >= 2 * 1024 * 8) return;
  int h = idx & 7;
  int n = idx >> 3;
  int b = n >> 10;
  int l = n & 1023;
  int bh = b * 8 + h;
  const float* Orow = O + ((size_t)bh * 1024 + l) * 80;
  for (int d = 0; d < 64; ++d)
    Y[(size_t)n * 608 + h * 64 + d] = f2bf(Orow[d]);
  const float* Rm = R + (size_t)n * 9;
  const float* t  = tr + (size_t)n * 3;
  for (int p = 0; p < 4; ++p) {
    float va[3];
#pragma unroll
    for (int x = 0; x < 3; ++x) va[x] = Orow[64 + p * 3 + x] - t[x];
#pragma unroll
    for (int x = 0; x < 3; ++x) {   // out[x] = sum_y R[y][x] * v[y]  (R^T)
      float vl = Rm[0 * 3 + x] * va[0] + Rm[1 * 3 + x] * va[1] + Rm[2 * 3 + x] * va[2];
      Y[(size_t)n * 608 + 512 + h * 12 + p * 3 + x] = f2bf(vl);
    }
  }
}

// ------------------------------------------- residual + bias + LayerNorm
__global__ __launch_bounds__(256) void ln_kernel(
    const float* __restrict__ s, const float* __restrict__ U,
    const float* __restrict__ bo, const float* __restrict__ gamma,
    const float* __restrict__ beta, float* __restrict__ out) {
  __shared__ float red[256];
  __shared__ float red2[256];
  int n = blockIdx.x;
  int tid = threadIdx.x;
  float x0 = s[(size_t)n * 512 + tid]       + U[(size_t)n * 512 + tid]       + bo[tid];
  float x1 = s[(size_t)n * 512 + 256 + tid] + U[(size_t)n * 512 + 256 + tid] + bo[256 + tid];
  red[tid]  = x0 + x1;
  red2[tid] = x0 * x0 + x1 * x1;
  __syncthreads();
  for (int off = 128; off > 0; off >>= 1) {
    if (tid < off) { red[tid] += red[tid + off]; red2[tid] += red2[tid + off]; }
    __syncthreads();
  }
  float mu  = red[0] * (1.f / 512.f);
  float var = red2[0] * (1.f / 512.f) - mu * mu;
  float rs  = rsqrtf(var + 1e-5f);
  out[(size_t)n * 512 + tid]       = (x0 - mu) * rs * gamma[tid]       + beta[tid];
  out[(size_t)n * 512 + 256 + tid] = (x1 - mu) * rs * gamma[256 + tid] + beta[256 + tid];
}

// --------------------------------------------------------------- launcher
extern "C" void kernel_launch(void* const* d_in, const int* in_sizes, int n_in,
                              void* d_out, int out_size, void* d_ws, size_t ws_size,
                              hipStream_t stream) {
  const float* s    = (const float*)d_in[0];
  const float* R    = (const float*)d_in[1];
  const float* tr   = (const float*)d_in[2];
  const float* Wq   = (const float*)d_in[3];
  const float* Wk   = (const float*)d_in[4];
  const float* Wv   = (const float*)d_in[5];
  const float* Wqp  = (const float*)d_in[6];
  const float* Wkp  = (const float*)d_in[7];
  const float* Wvp  = (const float*)d_in[8];
  const float* Wo   = (const float*)d_in[9];
  const float* bo   = (const float*)d_in[10];
  const float* gm   = (const float*)d_in[11];
  const float* bt   = (const float*)d_in[12];
  const float* wC   = (const float*)d_in[13];

  char* ws = (char*)d_ws;
  size_t off = 0;
  auto alloc = [&](size_t bytes) {
    void* p = ws + off;
    off = (off + bytes + 255) & ~(size_t)255;
    return p;
  };
  __bf16* s_bf   = (__bf16*)alloc(2048ull * 512 * 2);
  __bf16* Wq_bf  = (__bf16*)alloc(512ull * 512 * 2);
  __bf16* Wk_bf  = (__bf16*)alloc(512ull * 512 * 2);
  __bf16* Wv_bf  = (__bf16*)alloc(512ull * 512 * 2);
  __bf16* Wqp_bf = (__bf16*)alloc(512ull * 96 * 2);
  __bf16* Wkp_bf = (__bf16*)alloc(512ull * 96 * 2);
  __bf16* Wvp_bf = (__bf16*)alloc(512ull * 96 * 2);
  __bf16* Wo_bf  = (__bf16*)alloc(608ull * 512 * 2);
  float*  Qf     = (float*)alloc(2048ull * 512 * 4);
  float*  Kf     = (float*)alloc(2048ull * 512 * 4);
  float*  Vf     = (float*)alloc(2048ull * 512 * 4);
  float*  Qpf    = (float*)alloc(2048ull * 96 * 4);
  float*  Kpf    = (float*)alloc(2048ull * 96 * 4);
  float*  Vpf    = (float*)alloc(2048ull * 96 * 4);
  __bf16* Qe     = (__bf16*)alloc(16ull * 1024 * 96 * 2);
  __bf16* Ke     = (__bf16*)alloc(16ull * 1024 * 96 * 2);
  __bf16* VeT    = (__bf16*)alloc(16ull * 80 * 1024 * 2);
  float*  Kbias  = (float*)alloc(16ull * 1024 * 4);
  float*  O      = (float*)alloc(16ull * 1024 * 80 * 4);
  __bf16* Y      = (__bf16*)alloc(2048ull * 608 * 2);
  float*  U      = (float*)alloc(2048ull * 512 * 4);

  auto cvt = [&](const float* in, __bf16* out, int n) {
    int n4 = n / 4;
    cvt_f32_bf16_v4<<<(n4 + 255) / 256, 256, 0, stream>>>(
        (const float4*)in, (uint2*)out, n4);
  };
  cvt(s,   s_bf,   2048 * 512);
  cvt(Wq,  Wq_bf,  512 * 512);
  cvt(Wk,  Wk_bf,  512 * 512);
  cvt(Wv,  Wv_bf,  512 * 512);
  cvt(Wqp, Wqp_bf, 512 * 96);
  cvt(Wkp, Wkp_bf, 512 * 96);
  cvt(Wvp, Wvp_bf, 512 * 96);
  cvt(Wo,  Wo_bf,  608 * 512);

  auto gemm = [&](const __bf16* A, const __bf16* B, float* C, int M, int N, int K) {
    gemm_bf16_kernel<<<dim3(M / 128, (N + 63) / 64), 256, 0, stream>>>(A, B, C, M, N, K);
  };
  gemm(s_bf, Wq_bf,  Qf,  2048, 512, 512);
  gemm(s_bf, Wk_bf,  Kf,  2048, 512, 512);
  gemm(s_bf, Wv_bf,  Vf,  2048, 512, 512);
  gemm(s_bf, Wqp_bf, Qpf, 2048, 96,  512);
  gemm(s_bf, Wkp_bf, Kpf, 2048, 96,  512);
  gemm(s_bf, Wvp_bf, Vpf, 2048, 96,  512);

  point_setup_kernel<<<64, 256, 0, stream>>>(Qf, Kf, Vf, Qpf, Kpf, Vpf, R, tr, wC,
                                             Qe, Ke, VeT, Kbias);

  size_t attn_smem = 32ull * 1024 * 4 + 32ull * 96 * 2;   // 137,216 bytes
  attention_kernel<<<dim3(16, 32), 256, attn_smem, stream>>>(Qe, Ke, VeT, Kbias, O);

  build_y_kernel<<<64, 256, 0, stream>>>(O, R, tr, Y);

  gemm(Y, Wo_bf, U, 2048, 512, 608);

  ln_kernel<<<2048, 256, 0, stream>>>(s, U, bo, gm, bt, (float*)d_out);
}
